// ChildSumTreeLSTM_60086592471638
// MI455X (gfx1250) — compile-verified, matrix operand order checked by reading, not run on hardware
//
#include <hip/hip_runtime.h>
#include <hip/hip_bf16.h>

// ---------------------------------------------------------------------------
// Child-Sum Tree-LSTM on gfx1250 (MI455X), wave32 + WMMA bf16 (f32 accum).
//
// Phase 1: Y[N,608] = x[N,300] @ [W_ioux; W_fx]^T + b  (2x2-blocked WMMA GEMM)
// Phase 2: depth levels via parallel pointer jumping (17 fixed doubling
//          passes), then a persistent level-synchronous scan: each level is
//          batched 16-row WMMA GEMMs vs W_iouh^T / W_fh^T, sibling scatter
//          via global f32 atomics, grid barrier between levels.
// ---------------------------------------------------------------------------

typedef __attribute__((ext_vector_type(16))) __bf16 v16bf;
typedef __attribute__((ext_vector_type(8)))  float  v8f;

constexpr int kN      = 65536;     // nodes
constexpr int kInDim  = 300;
constexpr int kKpad   = 320;       // 300 -> 10 k-steps of 32
constexpr int kMem    = 150;
constexpr int kMemK   = 160;       // 150 -> 5 k-steps of 32
constexpr int kNOutP  = 608;       // 600 -> 38 column tiles of 16
constexpr int kNT1    = 38;
constexpr int kMT1    = kN / 16;   // 4096 row tiles
constexpr int kIouP   = 464;       // 450 -> 29 column tiles
constexpr int kNTiou  = 29;
constexpr int kNTf    = 10;        // 160 -> 10 column tiles
constexpr int kAccS   = 152;       // f32 row stride for h_acc / fc_acc
constexpr int kScanBlocks = 64;    // persistent cooperative grid
constexpr int kPJSteps    = 17;    // 2^17 > N: covers any parent chain

// --- WMMA fragment load: 16x32 bf16 tile, row-major source, stride in elems.
// Lane m+16h supplies row m; elems 0..7 -> K = base+8h.., elems 8..15 -> +16.
__device__ __forceinline__ v16bf load_frag(const __bf16* base, int stride,
                                           int row0, int col0) {
  int lane = threadIdx.x & 31;
  int m = lane & 15, h = lane >> 4;
  const __bf16* p = base + (size_t)(row0 + m) * stride + (col0 + 8 * h);
  union { uint4 u[2]; v16bf v; } f;
  f.u[0] = *reinterpret_cast<const uint4*>(p);
  f.u[1] = *reinterpret_cast<const uint4*>(p + 16);
  return f.v;
}

__device__ __forceinline__ float sigf(float x) { return 1.f / (1.f + __expf(-x)); }

// ---------------------------------------------------------------- prep / zero
__global__ void tlstm_zero_ws(unsigned int* p, size_t nWords) {
  size_t i = (size_t)blockIdx.x * blockDim.x + threadIdx.x;
  size_t stride = (size_t)gridDim.x * blockDim.x;
  for (; i < nWords; i += stride) p[i] = 0u;
}

__global__ void tlstm_prep_x(const float* __restrict__ x, __bf16* __restrict__ xb) {
  size_t idx = (size_t)blockIdx.x * blockDim.x + threadIdx.x;
  int i = (int)(idx / kKpad), k = (int)(idx % kKpad);
  float v = (k < kInDim) ? x[(size_t)i * kInDim + k] : 0.f;
  xb[idx] = (__bf16)v;
}

__global__ void tlstm_prep_w(const float* __restrict__ Wioux, const float* __restrict__ bioux,
                             const float* __restrict__ Wiouh, const float* __restrict__ Wfx,
                             const float* __restrict__ bfx,   const float* __restrict__ Wfh,
                             __bf16* __restrict__ WcatT, float* __restrict__ bcat,
                             __bf16* __restrict__ Wh1T,  __bf16* __restrict__ Wh2T) {
  int idx = blockIdx.x * blockDim.x + threadIdx.x;
  if (idx < kNOutP * kKpad) {                       // [608 x 320] = [Wioux;Wfx] rows
    int j = idx / kKpad, k = idx - j * kKpad;
    float v = 0.f;
    if (k < kInDim) {
      if (j < 450)      v = Wioux[j * kInDim + k];
      else if (j < 600) v = Wfx[(j - 450) * kInDim + k];
    }
    WcatT[idx] = (__bf16)v;  return;
  }
  idx -= kNOutP * kKpad;
  if (idx < kNOutP) {                               // fused bias
    float v = 0.f;
    if (idx < 450) v = bioux[idx]; else if (idx < 600) v = bfx[idx - 450];
    bcat[idx] = v;  return;
  }
  idx -= kNOutP;
  if (idx < kIouP * kMemK) {                        // [464 x 160] W_iouh rows
    int j = idx / kMemK, k = idx - j * kMemK;
    float v = (j < 450 && k < kMem) ? Wiouh[j * kMem + k] : 0.f;
    Wh1T[idx] = (__bf16)v;  return;
  }
  idx -= kIouP * kMemK;
  if (idx < kMemK * kMemK) {                        // [160 x 160] W_fh rows
    int j = idx / kMemK, k = idx - j * kMemK;
    float v = (j < kMem && k < kMem) ? Wfh[j * kMem + k] : 0.f;
    Wh2T[idx] = (__bf16)v;
  }
}

// ------------------------------------ phase 1: Y = x @ Wcat^T + b (2x2 tiles)
__global__ void tlstm_gemm_xw(const __bf16* __restrict__ xb, const __bf16* __restrict__ wT,
                              const float* __restrict__ bias, __bf16* __restrict__ Y) {
  constexpr int kTN2 = kNT1 / 2;                    // 19 column-pair tiles
  int wid = blockIdx.x * (blockDim.x >> 5) + (threadIdx.x >> 5);
  int tn2 = wid % kTN2, tm2 = wid / kTN2;
  int row0 = tm2 * 32, col0 = tn2 * 32;
  int lane = threadIdx.x & 31;
  v8f a00 = {}, a01 = {}, a10 = {}, a11 = {};
  for (int k = 0; k < kKpad; k += 32) {
    v16bf fa0 = load_frag(xb, kKpad, row0,      k);
    v16bf fa1 = load_frag(xb, kKpad, row0 + 16, k);
    v16bf fb0 = load_frag(wT, kKpad, col0,      k);
    v16bf fb1 = load_frag(wT, kKpad, col0 + 16, k);
    a00 = __builtin_amdgcn_wmma_f32_16x16x32_bf16(false, fa0, false, fb0, (short)0, a00, false, false);
    a01 = __builtin_amdgcn_wmma_f32_16x16x32_bf16(false, fa0, false, fb1, (short)0, a01, false, false);
    a10 = __builtin_amdgcn_wmma_f32_16x16x32_bf16(false, fa1, false, fb0, (short)0, a10, false, false);
    a11 = __builtin_amdgcn_wmma_f32_16x16x32_bf16(false, fa1, false, fb1, (short)0, a11, false, false);
  }
  int n = lane & 15, h = lane >> 4;
  float b0 = bias[col0 + n], b1 = bias[col0 + 16 + n];
#pragma unroll
  for (int r = 0; r < 8; ++r) {
    int m = r + 8 * h;
    Y[(size_t)(row0 + m) * kNOutP + col0 + n]           = (__bf16)(a00[r] + b0);
    Y[(size_t)(row0 + m) * kNOutP + col0 + 16 + n]      = (__bf16)(a01[r] + b1);
    Y[(size_t)(row0 + 16 + m) * kNOutP + col0 + n]      = (__bf16)(a10[r] + b0);
    Y[(size_t)(row0 + 16 + m) * kNOutP + col0 + 16 + n] = (__bf16)(a11[r] + b1);
  }
}

// ------------------------------- level schedule via parallel pointer jumping
// depth[i] = #edges from i to the virtual root slot N. Children are strictly
// deeper than parents, so processing levels in DESCENDING depth order is a
// valid children-first schedule. Level index L = maxDepth - depth[i].
__global__ void tlstm_pj_init(const int* __restrict__ parents,
                              int* __restrict__ anc, int* __restrict__ d) {
  int i = blockIdx.x * blockDim.x + threadIdx.x;
  if (i > kN) return;
  anc[i] = (i < kN) ? parents[i] : kN;
  d[i]   = (i < kN) ? 1 : 0;
}

__global__ void tlstm_pj_step(const int* __restrict__ anc, const int* __restrict__ d,
                              int* __restrict__ anc2, int* __restrict__ d2) {
  int i = blockIdx.x * blockDim.x + threadIdx.x;
  if (i > kN) return;
  int a = anc[i];
  d2[i]   = d[i] + d[a];
  anc2[i] = anc[a];
}

__global__ void tlstm_depth_max(const int* __restrict__ d, int* __restrict__ meta) {
  int i = blockIdx.x * blockDim.x + threadIdx.x;
  if (i >= kN) return;
  atomicMax(&meta[1], d[i]);
}

__global__ void tlstm_level_count(const int* __restrict__ d, const int* __restrict__ meta,
                                  int* __restrict__ levelStart) {
  int i = blockIdx.x * blockDim.x + threadIdx.x;
  if (i >= kN) return;
  atomicAdd(&levelStart[meta[1] - d[i]], 1);
}

__global__ void tlstm_prefix(int* __restrict__ levelStart, int* __restrict__ meta) {
  if (blockIdx.x != 0 || threadIdx.x != 0) return;
  int nl = meta[1];                 // = maxDepth; L in [0, nl-1]
  meta[0] = nl;
  int acc = 0;
  for (int l = 0; l < nl; ++l) { int c = levelStart[l]; levelStart[l] = acc; acc += c; }
  levelStart[nl] = acc;             // == kN
}

__global__ void tlstm_scatter(const int* __restrict__ d, const int* __restrict__ meta,
                              const int* __restrict__ levelStart, int* __restrict__ cur,
                              int* __restrict__ order) {
  int i = blockIdx.x * blockDim.x + threadIdx.x;
  if (i >= kN) return;
  int L = meta[1] - d[i];
  order[levelStart[L] + atomicAdd(&cur[L], 1)] = i;
}

// ------------------------------------------- phase 2: level-synchronous scan
__global__ void __launch_bounds__(256, 1)
tlstm_scan(const int* __restrict__ parents, const __bf16* __restrict__ Y,
           const __bf16* __restrict__ Wh1T, const __bf16* __restrict__ Wh2T,
           const float* __restrict__ b_iouh, const float* __restrict__ b_fh,
           float* __restrict__ h_acc, float* __restrict__ fc_acc,
           const int* __restrict__ order, const int* __restrict__ levelStart,
           const int* __restrict__ meta, int* __restrict__ bar,
           float* __restrict__ out) {
  __shared__ __align__(16) __bf16 sH[16 * kMemK];   // child-sum h, then h (bf16 A tile)
  __shared__ __align__(16) float  sF[16 * kIouP];   // iou pre-act, then f pre-act
  __shared__ float sC[16 * kAccS];                  // c (f32)
  __shared__ float sHh[16 * kAccS];                 // h (f32)
  __shared__ int   sNode[16];

  const int tid = threadIdx.x;
  const int lane = tid & 31, wave = tid >> 5;
  const int numLevels = meta[0];

  for (int lv = 0; lv < numLevels; ++lv) {
    const int s = levelStart[lv], e = levelStart[lv + 1];
    const int nTiles = (e - s + 15) >> 4;

    for (int t = blockIdx.x; t < nTiles; t += gridDim.x) {
      const int base = s + t * 16;
      if (tid < 16) sNode[tid] = (base + tid < e) ? order[base + tid] : -1;
      __syncthreads();

      // gather child-sum h into bf16 A tile (zero-padded)
      for (int idx = tid; idx < 16 * kMemK; idx += 256) {
        int r = idx / kMemK, k = idx - r * kMemK;
        int nd = sNode[r];
        float v = (nd >= 0 && k < kMem) ? h_acc[(size_t)nd * kAccS + k] : 0.f;
        sH[idx] = (__bf16)v;
      }
      __syncthreads();

      // GEMM1: [16x160] x [160x464] -> iou pre-activation (+ bias + iou_x).
      // A fragments preloaded once (5 k-steps x 8 VGPRs), reused for all ct.
      {
        v16bf aF[5];
#pragma unroll
        for (int ks = 0; ks < 5; ++ks) aF[ks] = load_frag(sH, kMemK, 0, ks * 32);
        for (int ct = wave; ct < kNTiou; ct += 8) {
          v8f acc = {};
#pragma unroll
          for (int ks = 0; ks < 5; ++ks) {
            v16bf b = load_frag(Wh1T, kMemK, ct * 16, ks * 32);
            acc = __builtin_amdgcn_wmma_f32_16x16x32_bf16(false, aF[ks], false, b,
                                                          (short)0, acc, false, false);
          }
          int n = lane & 15, hh = lane >> 4;
          int j = ct * 16 + n;
#pragma unroll
          for (int r = 0; r < 8; ++r) {
            int m = r + 8 * hh;
            float v = acc[r];
            if (j < 450) {
              int nd = sNode[m];
              float ioux = (nd >= 0) ? (float)Y[(size_t)nd * kNOutP + j] : 0.f;
              v += b_iouh[j] + ioux;
            }
            sF[m * kIouP + j] = v;
          }
        }
      }
      __syncthreads();

      // gates: c = sig(i)*tanh(u) + fc_sum ; h = sig(o)*tanh(c) ; emit h
      for (int idx = tid; idx < 16 * kMem; idx += 256) {
        int r = idx / kMem, j = idx - r * kMem;
        int nd = sNode[r];
        float ig = sF[r * kIouP + j];
        float og = sF[r * kIouP + kMem + j];
        float ug = sF[r * kIouP + 2 * kMem + j];
        float fc = (nd >= 0) ? fc_acc[(size_t)nd * kAccS + j] : 0.f;
        float c = sigf(ig) * tanhf(ug) + fc;
        float h = sigf(og) * tanhf(c);
        sC[r * kAccS + j] = c;
        sHh[r * kAccS + j] = h;
        sH[r * kMemK + j] = (__bf16)h;              // A tile for GEMM2
        if (nd >= 0) out[(size_t)nd * kMem + j] = h;
      }
      for (int idx = tid; idx < 16 * (kMemK - kMem); idx += 256) {
        int r = idx / (kMemK - kMem), k = kMem + (idx - r * (kMemK - kMem));
        sH[r * kMemK + k] = (__bf16)0.f;
      }
      __syncthreads();

      // GEMM2: [16x160] x [160x160] -> forget-gate pre-activation
      {
        v16bf aF[5];
#pragma unroll
        for (int ks = 0; ks < 5; ++ks) aF[ks] = load_frag(sH, kMemK, 0, ks * 32);
        for (int ct = wave; ct < kNTf; ct += 8) {
          v8f acc = {};
#pragma unroll
          for (int ks = 0; ks < 5; ++ks) {
            v16bf b = load_frag(Wh2T, kMemK, ct * 16, ks * 32);
            acc = __builtin_amdgcn_wmma_f32_16x16x32_bf16(false, aF[ks], false, b,
                                                          (short)0, acc, false, false);
          }
          int n = lane & 15, hh = lane >> 4;
          int j = ct * 16 + n;
#pragma unroll
          for (int r = 0; r < 8; ++r) sF[(r + 8 * hh) * kMemK + j] = acc[r];
        }
      }
      __syncthreads();

      // f = sig(Wfh·h + b_fh + fx[parent]); scatter h and f*c to parent slots
      for (int idx = tid; idx < 16 * kMem; idx += 256) {
        int r = idx / kMem, j = idx - r * kMem;
        int nd = sNode[r];
        if (nd < 0) continue;
        int p = parents[nd];
        float fxp = (p < kN) ? (float)Y[(size_t)p * kNOutP + 450 + j] : 0.f;
        float f = sigf(sF[r * kMemK + j] + b_fh[j] + fxp);
        atomicAdd(&h_acc[(size_t)p * kAccS + j], sHh[r * kAccS + j]);
        atomicAdd(&fc_acc[(size_t)p * kAccS + j], f * sC[r * kAccS + j]);
      }
      __syncthreads();
    }

    // grid-wide barrier between levels (monotonic counter, no reset needed)
    __syncthreads();
    if (tid == 0) {
      __threadfence();
      atomicAdd(bar, 1);
      const int target = (lv + 1) * (int)gridDim.x;
      while (__hip_atomic_load(bar, __ATOMIC_RELAXED, __HIP_MEMORY_SCOPE_AGENT) < target) {
        __builtin_amdgcn_s_sleep(2);
      }
      __threadfence();
    }
    __syncthreads();
  }
}

// ---------------------------------------------------------------------------
extern "C" void kernel_launch(void* const* d_in, const int* in_sizes, int n_in,
                              void* d_out, int out_size, void* d_ws, size_t ws_size,
                              hipStream_t stream) {
  (void)in_sizes; (void)n_in; (void)out_size; (void)ws_size;
  const float* x       = (const float*)d_in[0];
  const float* Wioux   = (const float*)d_in[1];
  const float* bioux   = (const float*)d_in[2];
  const float* Wiouh   = (const float*)d_in[3];
  const float* biouh   = (const float*)d_in[4];
  const float* Wfx     = (const float*)d_in[5];
  const float* bfx     = (const float*)d_in[6];
  const float* Wfh     = (const float*)d_in[7];
  const float* bfh     = (const float*)d_in[8];
  const int*   parents = (const int*)d_in[9];
  float* out = (float*)d_out;

  char* ws = (char*)d_ws;
  auto al = [](size_t v) { return (v + 255) & ~(size_t)255; };
  size_t off = 0;
  __bf16* xb     = (__bf16*)(ws + off); off = al(off + (size_t)kN * kKpad * 2);
  __bf16* WcatT  = (__bf16*)(ws + off); off = al(off + (size_t)kNOutP * kKpad * 2);
  float*  bcat   = (float*) (ws + off); off = al(off + (size_t)kNOutP * 4);
  __bf16* Wh1T   = (__bf16*)(ws + off); off = al(off + (size_t)kIouP * kMemK * 2);
  __bf16* Wh2T   = (__bf16*)(ws + off); off = al(off + (size_t)kMemK * kMemK * 2);
  __bf16* Yb     = (__bf16*)(ws + off); off = al(off + (size_t)kN * kNOutP * 2);
  size_t zero0 = off;
  float* h_acc   = (float*)(ws + off);  off = al(off + (size_t)(kN + 1) * kAccS * 4);
  float* fc_acc  = (float*)(ws + off);  off = al(off + (size_t)(kN + 1) * kAccS * 4);
  int* levelStart= (int*)(ws + off);    off = al(off + (size_t)(kN + 2) * 4);
  int* cur       = (int*)(ws + off);    off = al(off + (size_t)(kN + 1) * 4);
  int* meta      = (int*)(ws + off);    off = al(off + 64);
  int* bar       = (int*)(ws + off);    off = al(off + 64);
  size_t zero1 = off;
  int* anc0      = (int*)(ws + off);    off = al(off + (size_t)(kN + 1) * 4);
  int* anc1      = (int*)(ws + off);    off = al(off + (size_t)(kN + 1) * 4);
  int* dep0      = (int*)(ws + off);    off = al(off + (size_t)(kN + 1) * 4);
  int* dep1      = (int*)(ws + off);    off = al(off + (size_t)(kN + 1) * 4);
  int* order     = (int*)(ws + off);    off = al(off + (size_t)kN * 4);

  const int gN1 = (kN + 1 + 255) / 256;   // grids covering N(+1) elements
  const int gN  = (kN + 255) / 256;

  // 1) zero accumulators + level structures + barrier counter (every call)
  tlstm_zero_ws<<<2048, 256, 0, stream>>>((unsigned int*)(ws + zero0),
                                          (zero1 - zero0) / 4);
  // 2) pack x (bf16, K padded to 320)
  tlstm_prep_x<<<(kN * kKpad) / 256, 256, 0, stream>>>(x, xb);
  // 3) pack weights (row-major per output channel == WMMA B fragments)
  {
    int total = kNOutP * kKpad + kNOutP + kIouP * kMemK + kMemK * kMemK;
    tlstm_prep_w<<<(total + 255) / 256, 256, 0, stream>>>(
        Wioux, bioux, Wiouh, Wfx, bfx, Wfh, WcatT, bcat, Wh1T, Wh2T);
  }
  // 4) Y = x @ [W_ioux; W_fx]^T + b   (65536 x 608 x 320 bf16 WMMA GEMM)
  tlstm_gemm_xw<<<(kMT1 / 2) * (kNT1 / 2) / 8, 256, 0, stream>>>(xb, WcatT, bcat, Yb);
  // 5) tree depths by pointer jumping (fixed 17 doubling passes), then
  //    bucket nodes into levels ordered by descending depth
  tlstm_pj_init<<<gN1, 256, 0, stream>>>(parents, anc0, dep0);
  {
    int *a = anc0, *d = dep0, *a2 = anc1, *d2 = dep1;
    for (int it = 0; it < kPJSteps; ++it) {
      tlstm_pj_step<<<gN1, 256, 0, stream>>>(a, d, a2, d2);
      int* t;
      t = a; a = a2; a2 = t;
      t = d; d = d2; d2 = t;
    }
    // kPJSteps is odd -> final depths live in dep1's original buffer == `a`? No:
    // after swaps, `d` points at the buffer holding the latest result.
    tlstm_depth_max<<<gN, 256, 0, stream>>>(d, meta);
    tlstm_level_count<<<gN, 256, 0, stream>>>(d, meta, levelStart);
    tlstm_prefix<<<1, 1, 0, stream>>>(levelStart, meta);
    tlstm_scatter<<<gN, 256, 0, stream>>>(d, meta, levelStart, cur, order);
  }
  // 6) persistent level-synchronous WMMA scan
  tlstm_scan<<<kScanBlocks, 256, 0, stream>>>(parents, Yb, Wh1T, Wh2T, biouh, bfh,
                                              h_acc, fc_acc, order, levelStart,
                                              meta, bar, out);
}
